// GAT_21706764714529
// MI455X (gfx1250) — matrix operand order repaired and checked by reading, not compiled, not run on hardware
//
#include <hip/hip_runtime.h>
#include <math.h>
#include <stdint.h>

typedef __attribute__((ext_vector_type(16))) _Float16    v16h;
typedef __attribute__((ext_vector_type(8)))  float       v8f;
typedef __attribute__((ext_vector_type(8)))  unsigned int v8u;

#define GAT_WAVES   4     // waves per block
#define GAT_EPW     8     // batch elements per wave
#define GAT_ROWS    80    // GAT_EPW * 10 rows (xa:5 + xn:5 per element) = 5 M-tiles
#define H1_STRIDE   66    // halves; odd uint stride -> LDS bank spread
#define H2_STRIDE   36
#define NEG_SLOPE   0.2f

// staging offsets (in f32) inside the regA union for async weight DMA
#define SW1_OFF     0        // Wt1: 64*125  = 8000 f32 (32000 B, 16B aligned)
#define SW2_OFF     8000     // Wt2: 32*64   = 2048 f32 (offset 32000 B, 16B aligned)
#define SW3_OFF     10048    // Wl : 13*320  = 4160 f32 (offset 40192 B, 16B aligned)

// ---------------------------------------------------------------------------
// Fully fused GAT: load x once, 3 WMMA GEMM stages + 2 attention stages + BN,
// everything resident in LDS. f16 matmul inputs, f32 accumulation.
// Weights are DMA'd into LDS with global_load_async_to_lds_b128 (ASYNCcnt).
// ---------------------------------------------------------------------------
__launch_bounds__(128, 1)
__global__ void gat_fused(const float* __restrict__ x,
                          const float* __restrict__ Wt1,
                          const float* __restrict__ a11, const float* __restrict__ a21,
                          const float* __restrict__ g1,  const float* __restrict__ b1,
                          const float* __restrict__ m1,  const float* __restrict__ v1,
                          const float* __restrict__ Wt2,
                          const float* __restrict__ a12, const float* __restrict__ a22,
                          const float* __restrict__ g2,  const float* __restrict__ b2,
                          const float* __restrict__ m2,  const float* __restrict__ v2,
                          const float* __restrict__ Wl,  const float* __restrict__ bl,
                          float* __restrict__ out)
{
    // ---- weight fragments (B matrices), pre-swizzled to WMMA lane layout ----
    // B frag (32x16 f16): lane L holds column n=L%16, k_local = (L&16) + h
    __shared__ __align__(32) _Float16 W1F[4 * 4 * 32 * 16];   // kt=4, nt=4  (128x64)
    __shared__ __align__(32) _Float16 W2F[2 * 2 * 32 * 16];   // kt=2, nt=2  (64x32)
    __shared__ __align__(32) _Float16 W3F[10 * 32 * 16];      // kt=10, nt=1 (320x16)
    __shared__ float a11L[64], a21L[64], a12L[32], a22L[32];
    __shared__ float sc1L[5], sh1L[5], sc2L[5], sh2L[5], blL[16];
    // ---- per-wave working set (also reused as weight staging before phase 1)
    // regA: staging (f32 weights) -> XH (80x128 f16) -> A2 (80x64) -> FEAT (8x320)
    __shared__ __align__(32) _Float16 regA[GAT_WAVES][GAT_ROWS * 128];
    // regB: H1 (80 x H1_STRIDE f16), then H2 (80 x H2_STRIDE f16)
    __shared__ __align__(32) _Float16 regB[GAT_WAVES][GAT_ROWS * H1_STRIDE];
    __shared__ float SS[GAT_WAVES][GAT_ROWS * 2];

    const int tid = threadIdx.x;
    float* stage = (float*)&regA[0][0];            // 56.8 KB staging inside regA

    // ========= phase 0a: async DMA of raw f32 weights into LDS staging ======
    {
        const unsigned sbase = (unsigned)(uintptr_t)stage;   // LDS byte offset
        for (int idx = tid; idx < 2000; idx += 128) {        // Wt1: 2000 float4
            unsigned loff = sbase + SW1_OFF * 4 + idx * 16;
            unsigned long long g = (unsigned long long)(uintptr_t)(Wt1 + idx * 4);
            asm volatile("global_load_async_to_lds_b128 %0, %1, off"
                         :: "v"(loff), "v"(g) : "memory");
        }
        for (int idx = tid; idx < 512; idx += 128) {         // Wt2: 512 float4
            unsigned loff = sbase + SW2_OFF * 4 + idx * 16;
            unsigned long long g = (unsigned long long)(uintptr_t)(Wt2 + idx * 4);
            asm volatile("global_load_async_to_lds_b128 %0, %1, off"
                         :: "v"(loff), "v"(g) : "memory");
        }
        for (int idx = tid; idx < 1040; idx += 128) {        // Wl: 1040 float4
            unsigned loff = sbase + SW3_OFF * 4 + idx * 16;
            unsigned long long g = (unsigned long long)(uintptr_t)(Wl + idx * 4);
            asm volatile("global_load_async_to_lds_b128 %0, %1, off"
                         :: "v"(loff), "v"(g) : "memory");
        }
        asm volatile("s_wait_asynccnt 0x0" ::: "memory");    // own DMAs done
    }
    __syncthreads();                                          // everyone's done

    // ========= phase 0b: swizzle weights (LDS f32 -> fragment-major f16) ====
    for (int idx = tid; idx < 4 * 4 * 32 * 16; idx += 128) {
        int h = idx & 15, lane = (idx >> 4) & 31, nt = (idx >> 9) & 3, kt = idx >> 11;
        int k = kt * 32 + (lane & 16) + h;          // K of padded 128
        int n = nt * 16 + (lane & 15);              // N of 64
        float v = (k < 125) ? stage[SW1_OFF + n * 125 + k] : 0.f;
        W1F[idx] = (_Float16)v;
    }
    for (int idx = tid; idx < 2 * 2 * 32 * 16; idx += 128) {
        int h = idx & 15, lane = (idx >> 4) & 31, nt = (idx >> 9) & 1, kt = (idx >> 10) & 1;
        int k = kt * 32 + (lane & 16) + h;          // K of 64
        int n = nt * 16 + (lane & 15);              // N of 32
        W2F[idx] = (_Float16)stage[SW2_OFF + n * 64 + k];
    }
    for (int idx = tid; idx < 10 * 32 * 16; idx += 128) {
        int h = idx & 15, lane = (idx >> 4) & 31, kt = idx >> 9;
        int k = kt * 32 + (lane & 16) + h;          // K of 320
        int n = lane & 15;                          // N of 16 (13 valid)
        float v = (n < 13) ? stage[SW3_OFF + n * 320 + k] : 0.f;
        W3F[idx] = (_Float16)v;
    }
    for (int i = tid; i < 64; i += 128) { a11L[i] = a11[i]; a21L[i] = a21[i]; }
    for (int i = tid; i < 32; i += 128) { a12L[i] = a12[i]; a22L[i] = a22[i]; }
    if (tid < 5) {
        float inv = rsqrtf(v1[tid] + 1e-5f), s = inv * g1[tid];
        sc1L[tid] = s; sh1L[tid] = b1[tid] - m1[tid] * s;
        inv = rsqrtf(v2[tid] + 1e-5f); s = inv * g2[tid];
        sc2L[tid] = s; sh2L[tid] = b2[tid] - m2[tid] * s;
    }
    if (tid < 13) blL[tid] = bl[tid];
    __syncthreads();

    const int wave = tid >> 5, lane = tid & 31;
    const int mrow = lane & 15, halfw = lane >> 4;     // WMMA lane decomposition
    _Float16* XH = regA[wave];
    _Float16* H1 = regB[wave];
    float*    ss = SS[wave];
    const long elemBase = (long)blockIdx.x * (GAT_WAVES * GAT_EPW) + wave * GAT_EPW;

    // ============ phase 1: load x tile -> f16 rows (80 x 128, K padded) ======
    // x is streamed exactly once across the grid (328 MB): non-temporal loads.
    for (int idx = lane; idx < GAT_ROWS * 128; idx += 32) {
        int r = idx >> 7, k = idx & 127;
        int e = r / 10, rr = r - e * 10, part = rr / 5, i = rr - part * 5;
        float v = 0.f;
        if (k < 125)
            v = __builtin_nontemporal_load(
                    x + ((elemBase + e) * 5 + i) * 250 + part * 125 + k);
        XH[idx] = (_Float16)v;
    }
    __syncthreads();

    // ============ phase 2: GEMM1  h1 = XH(80x128) * Wt1^T(128x64) ============
    {
        const v16h* W1v = (const v16h*)W1F;
        const unsigned int* XHu = (const unsigned int*)XH;   // uint stride 64 per row
        for (int mt = 0; mt < 5; ++mt) {
            int r = mt * 16 + mrow;
            v16h a[4];
            #pragma unroll
            for (int kt = 0; kt < 4; ++kt) {
                v8u au;
                #pragma unroll
                for (int v = 0; v < 8; ++v)   // A frag: K pair base = 2v+16*half
                    au[v] = XHu[r * 64 + kt * 16 + v + 8 * halfw];
                a[kt] = __builtin_bit_cast(v16h, au);
            }
            #pragma unroll
            for (int nt = 0; nt < 4; ++nt) {
                v8f acc = {};
                #pragma unroll
                for (int kt = 0; kt < 4; ++kt) {
                    v16h b = W1v[(kt * 4 + nt) * 32 + lane];
                    acc = __builtin_amdgcn_wmma_f32_16x16x32_f16(
                        false, a[kt], false, b, (short)0, acc, false, false);
                }
                #pragma unroll
                for (int v = 0; v < 8; ++v)   // D: row m = v + 8*half, col n = mrow
                    H1[(mt * 16 + v + 8 * halfw) * H1_STRIDE + nt * 16 + mrow] = (_Float16)acc[v];
            }
        }
    }
    __syncthreads();

    // ============ phase 3: attention-1 scores s1=h1.a11, s2=h1.a21 ===========
    for (int it = 0; it < 3; ++it) {
        int r = lane + 32 * it;
        if (r < GAT_ROWS) {
            float s1 = 0.f, s2 = 0.f;
            for (int c = 0; c < 64; ++c) {
                float hv = (float)H1[r * H1_STRIDE + c];
                s1 += hv * a11L[c]; s2 += hv * a21L[c];
            }
            ss[r * 2] = s1; ss[r * 2 + 1] = s2;
        }
    }
    __syncthreads();

    // ===== phase 4: attention-1 softmax(dim=i) + mix + BN1 + ReLU -> A2 ======
    _Float16* A2 = regA[wave];                 // 80 x 64, overwrites XH
    for (int it = 0; it < 3; ++it) {
        int r = lane + 32 * it;
        if (r < GAT_ROWS) {
            int g = r / 5, i = r - g * 5, base = g * 5;
            float s1j[5], s2j[5], w[5];
            #pragma unroll
            for (int j = 0; j < 5; ++j) { s1j[j] = ss[(base + j) * 2]; s2j[j] = ss[(base + j) * 2 + 1]; }
            #pragma unroll
            for (int j = 0; j < 5; ++j) {      // softmax over i for fixed column j
                float mx = -1e30f;
                #pragma unroll
                for (int ii = 0; ii < 5; ++ii) {
                    float e = s1j[ii] + s2j[j];
                    e = e > 0.f ? e : NEG_SLOPE * e;
                    mx = fmaxf(mx, e);
                }
                float den = 0.f, mine = 0.f;
                #pragma unroll
                for (int ii = 0; ii < 5; ++ii) {
                    float e = s1j[ii] + s2j[j];
                    e = e > 0.f ? e : NEG_SLOPE * e;
                    float ex = __expf(e - mx);
                    den += ex;
                    if (ii == i) mine = ex;
                }
                w[j] = mine / den;
            }
            float sc = sc1L[i], sh = sh1L[i];
            for (int c = 0; c < 64; ++c) {
                float o = 0.f;
                #pragma unroll
                for (int j = 0; j < 5; ++j) o += w[j] * (float)H1[(base + j) * H1_STRIDE + c];
                o = o * sc + sh;
                A2[r * 64 + c] = (_Float16)fmaxf(o, 0.f);
            }
        }
    }
    __syncthreads();

    // ============ phase 5: GEMM2  h2 = A2(80x64) * Wt2^T(64x32) ==============
    _Float16* H2 = regB[wave];                 // 80 x H2_STRIDE, overwrites H1
    {
        const v16h* W2v = (const v16h*)W2F;
        const unsigned int* A2u = (const unsigned int*)A2;   // uint stride 32 per row
        for (int mt = 0; mt < 5; ++mt) {
            int r = mt * 16 + mrow;
            v16h a[2];
            #pragma unroll
            for (int kt = 0; kt < 2; ++kt) {
                v8u au;
                #pragma unroll
                for (int v = 0; v < 8; ++v)
                    au[v] = A2u[r * 32 + kt * 16 + v + 8 * halfw];
                a[kt] = __builtin_bit_cast(v16h, au);
            }
            #pragma unroll
            for (int nt = 0; nt < 2; ++nt) {
                v8f acc = {};
                #pragma unroll
                for (int kt = 0; kt < 2; ++kt) {
                    v16h b = W2v[(kt * 2 + nt) * 32 + lane];
                    acc = __builtin_amdgcn_wmma_f32_16x16x32_f16(
                        false, a[kt], false, b, (short)0, acc, false, false);
                }
                #pragma unroll
                for (int v = 0; v < 8; ++v)
                    H2[(mt * 16 + v + 8 * halfw) * H2_STRIDE + nt * 16 + mrow] = (_Float16)acc[v];
            }
        }
    }
    __syncthreads();

    // ============ phase 6: attention-2 scores p=h.a12, q=h.a22 ===============
    for (int it = 0; it < 3; ++it) {
        int r = lane + 32 * it;
        if (r < GAT_ROWS) {
            float p = 0.f, q = 0.f;
            for (int c = 0; c < 32; ++c) {
                float hv = (float)H2[r * H2_STRIDE + c];
                p += hv * a12L[c]; q += hv * a22L[c];
            }
            ss[r * 2] = p; ss[r * 2 + 1] = q;
        }
    }
    __syncthreads();

    // == phase 7: attention-2 (cross a<->n) + BN2 + ReLU -> FEAT (8 x 320) ====
    _Float16* FE = regA[wave];                 // overwrites A2
    for (int it = 0; it < 3; ++it) {
        int r = lane + 32 * it;
        if (r < GAT_ROWS) {
            int g = r / 5, i = r - g * 5;
            int part = g & 1, eL = g >> 1;
            int go = g * 5, gx = (g ^ 1) * 5;  // s1 from own h, s2 from sibling h
            float s1j[5], s2j[5], w[5];
            #pragma unroll
            for (int j = 0; j < 5; ++j) { s1j[j] = ss[(go + j) * 2]; s2j[j] = ss[(gx + j) * 2 + 1]; }
            #pragma unroll
            for (int j = 0; j < 5; ++j) {
                float mx = -1e30f;
                #pragma unroll
                for (int ii = 0; ii < 5; ++ii) {
                    float e = s1j[ii] + s2j[j];
                    e = e > 0.f ? e : NEG_SLOPE * e;
                    mx = fmaxf(mx, e);
                }
                float den = 0.f, mine = 0.f;
                #pragma unroll
                for (int ii = 0; ii < 5; ++ii) {
                    float e = s1j[ii] + s2j[j];
                    e = e > 0.f ? e : NEG_SLOPE * e;
                    float ex = __expf(e - mx);
                    den += ex;
                    if (ii == i) mine = ex;
                }
                w[j] = mine / den;
            }
            float sc = sc2L[i], sh = sh2L[i];
            for (int c = 0; c < 32; ++c) {
                float o = 0.f;
                #pragma unroll
                for (int j = 0; j < 5; ++j) o += w[j] * (float)H2[(go + j) * H2_STRIDE + c];
                o = o * sc + sh;
                // feat layout: [ya(5x32) | yn(5x32)] per node -> k = i*64 + part*32 + c
                FE[eL * 320 + i * 64 + part * 32 + c] = (_Float16)fmaxf(o, 0.f);
            }
        }
    }
    __syncthreads();

    // ============ phase 8: GEMM3  out = FEAT(8x320) * Wl^T(320x13) + bl ======
    {
        const v16h* W3v = (const v16h*)W3F;
        const unsigned int* FEu = (const unsigned int*)FE;   // uint stride 160 per row
        v8f acc = {};
        #pragma unroll
        for (int kt = 0; kt < 10; ++kt) {
            v8u au;
            #pragma unroll
            for (int v = 0; v < 8; ++v) {
                unsigned int u = 0u;
                if (mrow < GAT_EPW)            // rows 8..15 of the M-tile are zero pad
                    u = FEu[mrow * 160 + kt * 16 + v + 8 * halfw];
                au[v] = u;
            }
            v16h a = __builtin_bit_cast(v16h, au);
            v16h b = W3v[kt * 32 + lane];
            acc = __builtin_amdgcn_wmma_f32_16x16x32_f16(
                false, a, false, b, (short)0, acc, false, false);
        }
        #pragma unroll
        for (int v = 0; v < 8; ++v) {
            int m = v + 8 * halfw;             // batch element within wave
            if (m < GAT_EPW && mrow < 13)
                __builtin_nontemporal_store(acc[v] + blL[mrow],
                                            out + (elemBase + m) * 13 + mrow);
        }
    }
}

extern "C" void kernel_launch(void* const* d_in, const int* in_sizes, int n_in,
                              void* d_out, int out_size, void* d_ws, size_t ws_size,
                              hipStream_t stream) {
    const float* x   = (const float*)d_in[0];
    const float* Wt1 = (const float*)d_in[1];
    const float* a11 = (const float*)d_in[2];
    const float* a21 = (const float*)d_in[3];
    const float* g1  = (const float*)d_in[4];
    const float* b1  = (const float*)d_in[5];
    const float* m1  = (const float*)d_in[6];
    const float* v1  = (const float*)d_in[7];
    const float* Wt2 = (const float*)d_in[8];
    const float* a12 = (const float*)d_in[9];
    const float* a22 = (const float*)d_in[10];
    const float* g2  = (const float*)d_in[11];
    const float* b2  = (const float*)d_in[12];
    const float* m2  = (const float*)d_in[13];
    const float* v2  = (const float*)d_in[14];
    const float* Wl  = (const float*)d_in[15];
    const float* bl  = (const float*)d_in[16];
    float* out = (float*)d_out;

    const int B = in_sizes[0] / (5 * 250);                  // 65536
    const int blocks = B / (GAT_WAVES * GAT_EPW);           // 2048
    gat_fused<<<dim3(blocks), dim3(128), 0, stream>>>(
        x, Wt1, a11, a21, g1, b1, m1, v1,
        Wt2, a12, a22, g2, b2, m2, v2, Wl, bl, out);
}
